// GPTBlock_87608742904618
// MI455X (gfx1250) — compile-verified, matrix operand order checked by reading, not run on hardware
//
#include <hip/hip_runtime.h>
#include <hip/hip_bf16.h>

typedef unsigned short     u16;
typedef unsigned int       u32;
typedef unsigned long long u64;

typedef __attribute__((ext_vector_type(16))) __bf16 v16bf;
typedef __attribute__((ext_vector_type(8)))  float  v8f;
typedef __attribute__((ext_vector_type(4)))  u32    v4u;
typedef __attribute__((ext_vector_type(8)))  int    v8i;
typedef __attribute__((ext_vector_type(4)))  int    v4i;

// ---------------- problem constants ----------------
#define Bb   2
#define Ss   2048
#define Dd   2048
#define Hh   16
#define DKk  128
#define Mm   (Bb * Ss)      // 4096 rows
#define D4   (4 * Dd)       // 8192

// ---------------- helpers ----------------
__device__ __forceinline__ u16 f2bf(float f) {
    u32 u = __float_as_uint(f);
    u32 r = (u + 0x7FFFu + ((u >> 16) & 1u)) >> 16;   // round-to-nearest-even
    return (u16)r;
}
__device__ __forceinline__ float bf2f(u16 h) {
    return __uint_as_float(((u32)h) << 16);
}

// Load one 16x32 bf16 fragment half-pattern: lane reads 8 contiguous K at p
// and 8 more at p+16 (matches CDNA5 16-bit A/B per-lane layout).
__device__ __forceinline__ v16bf ld_frag(const u16* p) {
    union { uint4 q[2]; v16bf v; } u;
    u.q[0] = *(const uint4*)(p);
    u.q[1] = *(const uint4*)(p + 16);
    return u.v;
}

__device__ __forceinline__ v8f wmma_bf16(v16bf a, v16bf b, v8f c) {
    return __builtin_amdgcn_wmma_f32_16x16x32_bf16(
        /*neg_a=*/false, a, /*neg_b=*/false, b,
        /*c_mod=*/(short)0, c, /*reuse_a=*/false, /*reuse_b=*/false);
}

__device__ __forceinline__ float gelu_exact(float v) {
    return 0.5f * v * (1.0f + erff(v * 0.70710678118654752f));
}

// ---------------- TDM: async tensor load of a 2D bf16 tile into LDS ----------
// Loads a [tile_d1 rows x tile_d0 elems] bf16 tile whose rows are tensor_d0
// elems apart in global memory, into LDS at lds_off, inserting a 16B pad
// after every 64B row (pad_interval=3 -> every 8x8B; pad_amount=3 -> 4 DWORDs)
// so the LDS row stride is 80B = LDT bf16 elems.
__device__ __forceinline__ void tdm_tile_load(u32 lds_off, const u16* gsrc,
                                              u32 tensor_d0, u32 tensor_d1,
                                              u32 tile_d0, u32 tile_d1) {
    u64 ga = (u64)(uintptr_t)gsrc;
    v4u g0;
    g0.x = 1u;                                          // count=1, user mode
    g0.y = lds_off;                                     // LDS byte address
    g0.z = (u32)ga;                                     // global addr [31:0]
    g0.w = ((u32)(ga >> 32) & 0x01FFFFFFu) | (2u << 30); // addr[56:32] | type=2
    v8i g1;
    g1[0] = (int)((1u << 16) | (1u << 20) | (3u << 22) | (3u << 25));
            // data_size=1 (2B) | pad_enable | pad_interval=3 | pad_amount=3
    g1[1] = (int)((tensor_d0 & 0xFFFFu) << 16);          // tensor_dim0 lo
    g1[2] = (int)((tensor_d0 >> 16) | ((tensor_d1 & 0xFFFFu) << 16));
    g1[3] = (int)((tensor_d1 >> 16) | (tile_d0 << 16));  // | tile_dim0
    g1[4] = (int)tile_d1;                                // tile_dim1 (dim2=0)
    g1[5] = (int)tensor_d0;                              // dim0 stride [31:0]
    g1[6] = 0;                                           // stride hi | dim1 stride lo
    g1[7] = 0;
    v4i z4 = {0, 0, 0, 0};
    v8i z8 = {0, 0, 0, 0, 0, 0, 0, 0};
    __builtin_amdgcn_tensor_load_to_lds(g0, g1, z4, z4, z8, 0);
}

// ---------------- weight transpose + fp32->bf16 ----------------
// in: [Kd][Nd] fp32 (row-major)   out: [Nd][Kd] bf16 (K contiguous)
__global__ __launch_bounds__(256) void wt_kernel(const float* __restrict__ in,
                                                 u16* __restrict__ out,
                                                 int Kd, int Nd) {
    __shared__ float tile[32][33];
    int nx = blockIdx.x * 32;
    int kx = blockIdx.y * 32;
    int tx = threadIdx.x;   // 0..31
    int ty = threadIdx.y;   // 0..7
#pragma unroll
    for (int i = 0; i < 4; ++i)
        tile[ty + 8 * i][tx] = in[(size_t)(kx + ty + 8 * i) * Nd + nx + tx];
    __syncthreads();
#pragma unroll
    for (int i = 0; i < 4; ++i)
        out[(size_t)(nx + ty + 8 * i) * Kd + kx + tx] = f2bf(tile[tx][ty + 8 * i]);
}

// ---------------- layernorm fp32 -> bf16 ----------------
__global__ __launch_bounds__(256) void ln_kernel(const float* __restrict__ x,
                                                 const float* __restrict__ g,
                                                 const float* __restrict__ bta,
                                                 u16* __restrict__ out) {
    __shared__ float red[256], red2[256];
    int row = blockIdx.x;
    int tid = threadIdx.x;
    const float* xr = x + (size_t)row * Dd;
    float s = 0.f, s2 = 0.f;
    for (int i = tid; i < Dd; i += 256) { float v = xr[i]; s += v; s2 += v * v; }
    red[tid] = s; red2[tid] = s2;
    __syncthreads();
    for (int off = 128; off > 0; off >>= 1) {
        if (tid < off) { red[tid] += red[tid + off]; red2[tid] += red2[tid + off]; }
        __syncthreads();
    }
    float mean = red[0] * (1.0f / Dd);
    float var  = red2[0] * (1.0f / Dd) - mean * mean;
    float inv  = rsqrtf(var + 1e-5f);
    for (int i = tid; i < Dd; i += 256) {
        float v = (xr[i] - mean) * inv * g[i] + bta[i];
        out[(size_t)row * Dd + i] = f2bf(v);
    }
}

// ---------------- bf16 WMMA GEMM:  C = A[M,K] x Bt[N,K]^T + bias ----------------
// TDM double-buffered: wave 0 issues tensor_load_to_lds for the next K slice
// while all 8 waves compute WMMAs on the current buffer.
enum { EP_BF16 = 0, EP_BF16_VT = 1, EP_BF16_GELU = 2, EP_F32_RES = 3 };

#define TM 128
#define TN 128
#define TK 32
#define LDT (TK + 8)   // bf16 elements; 80B row stride (matches TDM padding)

template <int EP>
__global__ __launch_bounds__(256) void gemm_kernel(const u16* __restrict__ A,
                                                   const u16* __restrict__ Bt,
                                                   const float* __restrict__ bias,
                                                   const float* __restrict__ res,
                                                   void* __restrict__ outv,
                                                   int Mn, int Nn, int Kn) {
    __shared__ alignas(16) u16 As[2][TM * LDT];
    __shared__ alignas(16) u16 Bs[2][TN * LDT];

    const int tid  = threadIdx.x;
    const int lane = tid & 31;
    const int wv   = tid >> 5;
    const int mw   = wv >> 1;          // 0..3  -> 32 rows each
    const int nw   = wv & 1;           // 0..1  -> 64 cols each
    const int m0   = blockIdx.y * TM;
    const int n0   = blockIdx.x * TN;
    const int half = lane >> 4;
    const int l16  = lane & 15;

    v8f acc[2][4];
#pragma unroll
    for (int i = 0; i < 2; ++i)
#pragma unroll
        for (int j = 0; j < 4; ++j) acc[i][j] = (v8f){0,0,0,0,0,0,0,0};

    const u16* gA0 = A  + (size_t)m0 * Kn;   // tile row 0 of this block
    const u16* gB0 = Bt + (size_t)n0 * Kn;
    const u32 aoff[2] = { (u32)(uintptr_t)&As[0][0], (u32)(uintptr_t)&As[1][0] };
    const u32 boff[2] = { (u32)(uintptr_t)&Bs[0][0], (u32)(uintptr_t)&Bs[1][0] };
    const bool issuer = (wv == 0);

    if (issuer) {   // prologue: fill buffer 0
        tdm_tile_load(aoff[0], gA0, (u32)Kn, (u32)Mn, TK, TM);
        tdm_tile_load(boff[0], gB0, (u32)Kn, (u32)Nn, TK, TN);
    }

    const int nIter = Kn / TK;
    for (int it = 0; it < nIter; ++it) {
        const int cur = it & 1;
        if (issuer) {
            if (it + 1 < nIter) {   // issue next slice into the other buffer
                const int kn = (it + 1) * TK;
                tdm_tile_load(aoff[cur ^ 1], gA0 + kn, (u32)Kn, (u32)Mn, TK, TM);
                tdm_tile_load(boff[cur ^ 1], gB0 + kn, (u32)Kn, (u32)Nn, TK, TN);
                // TDM completes in order: <=2 outstanding => current pair done
                __builtin_amdgcn_s_wait_tensorcnt(2);
            } else {
                __builtin_amdgcn_s_wait_tensorcnt(0);
            }
        }
        __syncthreads();   // current buffer visible to all waves

        v16bf af[2], bfr[4];
#pragma unroll
        for (int mt = 0; mt < 2; ++mt) {
            int r = mw * 32 + mt * 16 + l16;
            af[mt] = ld_frag(&As[cur][r * LDT + half * 8]);
        }
#pragma unroll
        for (int nt = 0; nt < 4; ++nt) {
            int r = nw * 64 + nt * 16 + l16;
            bfr[nt] = ld_frag(&Bs[cur][r * LDT + half * 8]);
        }
#pragma unroll
        for (int mt = 0; mt < 2; ++mt)
#pragma unroll
            for (int nt = 0; nt < 4; ++nt)
                acc[mt][nt] = wmma_bf16(af[mt], bfr[nt], acc[mt][nt]);
        __syncthreads();   // all waves done reading before buffer is reused
    }

    // ---- epilogue ----
    float bv[4];
#pragma unroll
    for (int nt = 0; nt < 4; ++nt)
        bv[nt] = bias[n0 + nw * 64 + nt * 16 + l16];

#pragma unroll
    for (int mt = 0; mt < 2; ++mt)
#pragma unroll
        for (int nt = 0; nt < 4; ++nt)
#pragma unroll
            for (int j = 0; j < 8; ++j) {
                int row = m0 + mw * 32 + mt * 16 + j + half * 8;
                int col = n0 + nw * 64 + nt * 16 + l16;
                float v = acc[mt][nt][j] + bv[nt];
                if (EP == EP_BF16_GELU) v = gelu_exact(v);
                if (EP == EP_F32_RES) {
                    float* o = (float*)outv;
                    o[(size_t)row * Nn + col] = v + res[(size_t)row * Nn + col];
                } else if (EP == EP_BF16_VT) {
                    // scatter as V^T: [b, h, dk, s]
                    int bb = row / Ss, ss = row % Ss;
                    int hh = col / DKk, dk = col % DKk;
                    u16* o = (u16*)outv;
                    o[(((size_t)bb * Hh + hh) * DKk + dk) * Ss + ss] = f2bf(v);
                } else {
                    u16* o = (u16*)outv;
                    o[(size_t)row * Nn + col] = f2bf(v);
                }
            }
}

// ---------------- flash attention (bf16 WMMA, online softmax) ----------------
// Grid: (S/32, H, B). 256 threads = 8 waves.
// Wave w: phase A computes score keys [w*32, w*32+32); phase C owns dk slice w*16.
__global__ __launch_bounds__(256) void attn_kernel(const u16* __restrict__ Q,
                                                   const u16* __restrict__ Km,
                                                   const u16* __restrict__ Vt,
                                                   u16* __restrict__ Ctx) {
    constexpr int KB   = 256;       // keys per block
    constexpr int SSTR = KB + 8;    // float stride
    constexpr int PSTR = KB + 8;    // bf16 stride (528B, 16B aligned)
    __shared__ float sS[32 * SSTR];
    __shared__ alignas(16) u16 sP[32 * PSTR];
    __shared__ float sCorr[32];
    __shared__ float sSum[32];

    const int b  = blockIdx.z, h = blockIdx.y;
    const int q0 = blockIdx.x * 32;
    const int tid = threadIdx.x, lane = tid & 31, wv = tid >> 5;
    const int half = lane >> 4, l16 = lane & 15;
    const float scale = 0.088388347648318447f;   // 1/sqrt(128)

    v8f acc0 = (v8f){0,0,0,0,0,0,0,0};
    v8f acc1 = (v8f){0,0,0,0,0,0,0,0};
    float rmax = -__builtin_inff();
    float rsum = 0.f;

    const size_t qbase = (size_t)b * Ss * Dd + (size_t)h * DKk;
    const int nkb = q0 / KB + 1;    // causal: only key blocks with k0 <= q0

    for (int kb = 0; kb < nkb; ++kb) {
        const int k0  = kb * KB;
        const int kw0 = k0 + wv * 32;

        // ---- phase A: S = Q K^T for this wave's 32 keys ----
        v8f sc[2][2];
#pragma unroll
        for (int i = 0; i < 2; ++i)
#pragma unroll
            for (int j = 0; j < 2; ++j) sc[i][j] = (v8f){0,0,0,0,0,0,0,0};
#pragma unroll
        for (int c = 0; c < 4; ++c) {            // DK=128 in 4 chunks of 32
            const int dko = c * 32 + half * 8;
            v16bf a0 = ld_frag(Q  + qbase + (size_t)(q0 + l16)       * Dd + dko);
            v16bf a1 = ld_frag(Q  + qbase + (size_t)(q0 + 16 + l16)  * Dd + dko);
            v16bf b0 = ld_frag(Km + qbase + (size_t)(kw0 + l16)      * Dd + dko);
            v16bf b1 = ld_frag(Km + qbase + (size_t)(kw0 + 16 + l16) * Dd + dko);
            sc[0][0] = wmma_bf16(a0, b0, sc[0][0]);
            sc[0][1] = wmma_bf16(a0, b1, sc[0][1]);
            sc[1][0] = wmma_bf16(a1, b0, sc[1][0]);
            sc[1][1] = wmma_bf16(a1, b1, sc[1][1]);
        }
        // mask + scale, stage into LDS (C layout -> [row][key])
#pragma unroll
        for (int mt = 0; mt < 2; ++mt)
#pragma unroll
            for (int nt = 0; nt < 2; ++nt)
#pragma unroll
                for (int j = 0; j < 8; ++j) {
                    int rloc = mt * 16 + j + half * 8;
                    int kj   = kw0 + nt * 16 + l16;
                    int qi   = q0 + rloc;
                    float v  = (kj <= qi) ? sc[mt][nt][j] * scale : -1e9f;
                    sS[rloc * SSTR + wv * 32 + nt * 16 + l16] = v;
                }
        __syncthreads();

        // ---- phase B: online softmax, one thread per query row ----
        if (tid < 32) {
            int r = tid;
            float bmax = -1e30f;
            for (int j = 0; j < KB; ++j) bmax = fmaxf(bmax, sS[r * SSTR + j]);
            float mnew = fmaxf(rmax, bmax);
            float corr = __expf(rmax - mnew);     // first block: exp(-inf)=0
            rsum *= corr;
            const float* srow = &sS[r * SSTR];
            u16* prow = &sP[r * PSTR];
            for (int j = 0; j < KB; ++j) {
                float p = __expf(srow[j] - mnew);
                rsum += p;
                prow[j] = f2bf(p);
            }
            sCorr[r] = corr;
            sSum[r]  = rsum;
            rmax = mnew;
        }
        __syncthreads();

        // ---- phase C: ctx = ctx*corr + P @ V  (wave owns dk slice wv*16) ----
#pragma unroll
        for (int j = 0; j < 8; ++j) {
            acc0[j] *= sCorr[j + half * 8];
            acc1[j] *= sCorr[16 + j + half * 8];
        }
        const size_t vtb = (((size_t)b * Hh + h) * DKk + wv * 16 + l16) * Ss;
#pragma unroll
        for (int c = 0; c < 8; ++c) {            // 256 keys in 8 chunks of 32
            const int ko = c * 32 + half * 8;
            v16bf p0 = ld_frag(&sP[l16 * PSTR + ko]);
            v16bf p1 = ld_frag(&sP[(16 + l16) * PSTR + ko]);
            v16bf bv = ld_frag(Vt + vtb + k0 + ko);
            acc0 = wmma_bf16(p0, bv, acc0);
            acc1 = wmma_bf16(p1, bv, acc1);
        }
        __syncthreads();
    }

    // ---- normalize + store ctx bf16 as [b, s, h*DK+dk] ----
#pragma unroll
    for (int j = 0; j < 8; ++j) {
        int r0 = j + half * 8;
        int r1 = 16 + r0;
        float i0 = 1.0f / sSum[r0];
        float i1 = 1.0f / sSum[r1];
        size_t o0 = (size_t)(b * Ss + q0 + r0) * Dd + h * DKk + wv * 16 + l16;
        size_t o1 = (size_t)(b * Ss + q0 + r1) * Dd + h * DKk + wv * 16 + l16;
        Ctx[o0] = f2bf(acc0[j] * i0);
        Ctx[o1] = f2bf(acc1[j] * i1);
    }
}

// ---------------- workspace layout ----------------
static constexpr size_t SZ_WDD  = (size_t)Dd * Dd * 2;     // bf16 DxD weight
static constexpr size_t SZ_WD4  = (size_t)Dd * D4 * 2;     // bf16 DxD4 weight
static constexpr size_t SZ_ACT  = (size_t)Mm * Dd * 2;     // bf16 activation
static constexpr size_t SZ_ACTF = (size_t)Mm * Dd * 4;     // fp32 activation
static constexpr size_t OFF_WQT  = 0;
static constexpr size_t OFF_WKT  = OFF_WQT + SZ_WDD;
static constexpr size_t OFF_WVT  = OFF_WKT + SZ_WDD;
static constexpr size_t OFF_WOT  = OFF_WVT + SZ_WDD;
static constexpr size_t OFF_W1T  = OFF_WOT + SZ_WDD;
static constexpr size_t OFF_W2T  = OFF_W1T + SZ_WD4;
static constexpr size_t OFF_XLN  = OFF_W2T + SZ_WD4;
static constexpr size_t OFF_Q    = OFF_XLN + SZ_ACT;
static constexpr size_t OFF_K    = OFF_Q + SZ_ACT;
static constexpr size_t OFF_VT   = OFF_K + SZ_ACT;
static constexpr size_t OFF_CTX  = OFF_VT + SZ_ACT;
static constexpr size_t OFF_XATT = OFF_CTX + SZ_ACT;
static constexpr size_t OFF_YLN  = OFF_XATT + SZ_ACTF;
static constexpr size_t OFF_H1   = OFF_YLN + SZ_ACT;

extern "C" void kernel_launch(void* const* d_in, const int* in_sizes, int n_in,
                              void* d_out, int out_size, void* d_ws, size_t ws_size,
                              hipStream_t stream) {
    (void)in_sizes; (void)n_in; (void)out_size; (void)ws_size;
    const float* src  = (const float*)d_in[0];
    // d_in[1] = src_mask (causal; hardcoded in attn_kernel)
    const float* Wq   = (const float*)d_in[2];
    const float* bq   = (const float*)d_in[3];
    const float* Wk   = (const float*)d_in[4];
    const float* bk   = (const float*)d_in[5];
    const float* Wv   = (const float*)d_in[6];
    const float* bv   = (const float*)d_in[7];
    const float* Wo   = (const float*)d_in[8];
    const float* bo   = (const float*)d_in[9];
    const float* ln1g = (const float*)d_in[10];
    const float* ln1b = (const float*)d_in[11];
    const float* ln2g = (const float*)d_in[12];
    const float* ln2b = (const float*)d_in[13];
    const float* W1   = (const float*)d_in[14];
    const float* b1   = (const float*)d_in[15];
    const float* W2   = (const float*)d_in[16];
    const float* b2   = (const float*)d_in[17];
    float* out = (float*)d_out;

    char* ws = (char*)d_ws;
    u16* wqt  = (u16*)(ws + OFF_WQT);
    u16* wkt  = (u16*)(ws + OFF_WKT);
    u16* wvt  = (u16*)(ws + OFF_WVT);
    u16* wot  = (u16*)(ws + OFF_WOT);
    u16* w1t  = (u16*)(ws + OFF_W1T);
    u16* w2t  = (u16*)(ws + OFF_W2T);
    u16* xln  = (u16*)(ws + OFF_XLN);
    u16* qb   = (u16*)(ws + OFF_Q);
    u16* kbuf = (u16*)(ws + OFF_K);
    u16* vt   = (u16*)(ws + OFF_VT);
    u16* ctx  = (u16*)(ws + OFF_CTX);
    float* xatt = (float*)(ws + OFF_XATT);
    u16* yln  = (u16*)(ws + OFF_YLN);
    u16* h1   = (u16*)(ws + OFF_H1);

    dim3 tb(32, 8);
    // weight transpose + bf16 convert
    wt_kernel<<<dim3(Dd / 32, Dd / 32), tb, 0, stream>>>(Wq, wqt, Dd, Dd);
    wt_kernel<<<dim3(Dd / 32, Dd / 32), tb, 0, stream>>>(Wk, wkt, Dd, Dd);
    wt_kernel<<<dim3(Dd / 32, Dd / 32), tb, 0, stream>>>(Wv, wvt, Dd, Dd);
    wt_kernel<<<dim3(Dd / 32, Dd / 32), tb, 0, stream>>>(Wo, wot, Dd, Dd);
    wt_kernel<<<dim3(D4 / 32, Dd / 32), tb, 0, stream>>>(W1, w1t, Dd, D4); // [D][D4]->[D4][D]
    wt_kernel<<<dim3(Dd / 32, D4 / 32), tb, 0, stream>>>(W2, w2t, D4, Dd); // [D4][D]->[D][D4]

    // LN1
    ln_kernel<<<Mm, 256, 0, stream>>>(src, ln1g, ln1b, xln);

    // Q, K projections (bf16 out), V projection scattered as V^T
    gemm_kernel<EP_BF16><<<dim3(Dd / TN, Mm / TM), 256, 0, stream>>>(
        xln, wqt, bq, nullptr, qb, Mm, Dd, Dd);
    gemm_kernel<EP_BF16><<<dim3(Dd / TN, Mm / TM), 256, 0, stream>>>(
        xln, wkt, bk, nullptr, kbuf, Mm, Dd, Dd);
    gemm_kernel<EP_BF16_VT><<<dim3(Dd / TN, Mm / TM), 256, 0, stream>>>(
        xln, wvt, bv, nullptr, vt, Mm, Dd, Dd);

    // flash attention
    attn_kernel<<<dim3(Ss / 32, Hh, Bb), 256, 0, stream>>>(qb, kbuf, vt, ctx);

    // output projection + residual (fp32)
    gemm_kernel<EP_F32_RES><<<dim3(Dd / TN, Mm / TM), 256, 0, stream>>>(
        ctx, wot, bo, src, xatt, Mm, Dd, Dd);

    // LN2
    ln_kernel<<<Mm, 256, 0, stream>>>(xatt, ln2g, ln2b, yln);

    // MLP: GELU(y W1 + b1) W2 + b2 + residual
    gemm_kernel<EP_BF16_GELU><<<dim3(D4 / TN, Mm / TM), 256, 0, stream>>>(
        yln, w1t, b1, nullptr, h1, Mm, D4, Dd);
    gemm_kernel<EP_F32_RES><<<dim3(Dd / TN, Mm / TM), 256, 0, stream>>>(
        h1, w2t, b2, xatt, out, Mm, Dd, D4);
}
